// BaselineModel_80607946211554
// MI455X (gfx1250) — matrix operand order repaired and checked by reading, not compiled, not run on hardware
//
#include <hip/hip_runtime.h>
#include <hip/hip_bf16.h>

typedef __bf16 bf16_t;
typedef bf16_t v16bf __attribute__((ext_vector_type(16)));
typedef bf16_t v8bf  __attribute__((ext_vector_type(8)));
typedef bf16_t v4bf  __attribute__((ext_vector_type(4)));
typedef float  v8f   __attribute__((ext_vector_type(8)));

#define N_NODES 50000
#define N_EDGES 1600000
#define N_GRAPH 256
#define HDIM    128
#define NCLS    10

// ---------------------------------------------------------------- utilities

__global__ __launch_bounds__(256) void k_f32_to_bf16(const float* __restrict__ in,
                                                     bf16_t* __restrict__ out) {
    unsigned i = blockIdx.x * 256u + threadIdx.x;   // exactly N*H threads launched
    out[i] = (bf16_t)in[i];
}

// Wt[n*128 + k] = W[k*128 + n]  (128x128 f32 -> transposed bf16)
__global__ __launch_bounds__(256) void k_transpose128(const float* __restrict__ W,
                                                      bf16_t* __restrict__ Wt) {
    unsigned i = blockIdx.x * 256u + threadIdx.x;   // 16384 threads
    unsigned n = i >> 7, k = i & 127u;
    Wt[n * 128u + k] = (bf16_t)W[k * 128u + n];
}

__global__ __launch_bounds__(256) void k_degree(const int* __restrict__ dst,
                                                float* __restrict__ deg) {
    unsigned e = blockIdx.x * 256u + threadIdx.x;   // exactly E threads
    unsafeAtomicAdd(&deg[dst[e]], 1.0f);
}

// 32 lanes per edge, 4 bf16 elements per lane: agg[dst] += act[src]
__global__ __launch_bounds__(256) void k_scatter(const bf16_t* __restrict__ act,
                                                 const int* __restrict__ src,
                                                 const int* __restrict__ dst,
                                                 float* __restrict__ agg) {
    unsigned idx = blockIdx.x * 256u + threadIdx.x; // E*32 threads
    unsigned e = idx >> 5;
    unsigned t = (idx & 31u) << 2;                  // element 0..124
    int s = src[e], d = dst[e];
    v4bf a = *(const v4bf*)(act + (size_t)s * HDIM + t);
    float* o = agg + (size_t)d * HDIM + t;
    unsafeAtomicAdd(o + 0, (float)a[0]);
    unsafeAtomicAdd(o + 1, (float)a[1]);
    unsafeAtomicAdd(o + 2, (float)a[2]);
    unsafeAtomicAdd(o + 3, (float)a[3]);
}

// mean_bf = bf16(agg / max(deg,1))
__global__ __launch_bounds__(256) void k_mean_cvt(const float* __restrict__ agg,
                                                  const float* __restrict__ deg,
                                                  bf16_t* __restrict__ mb) {
    unsigned i = blockIdx.x * 256u + threadIdx.x;   // N*H threads
    float d = fmaxf(deg[i >> 7], 1.0f);
    mb[i] = (bf16_t)(agg[i] / d);
}

// pool[batch[node], colbase + j] += hb[node, j]
__global__ __launch_bounds__(256) void k_pool(const bf16_t* __restrict__ hb,
                                              const int* __restrict__ batch,
                                              float* __restrict__ pool, int colbase) {
    unsigned i = blockIdx.x * 256u + threadIdx.x;   // N*H threads
    unsigned node = i >> 7, j = i & 127u;
    int g = batch[node];
    unsafeAtomicAdd(&pool[(size_t)g * (2 * HDIM) + colbase + j], (float)hb[i]);
}

// ---------------------------------------------------------------- WMMA GEMM
// out[N,128] = relu( A1[N,128] @ W1 + A2[N,128] @ W2 + bias )   (bf16 in/out, f32 acc)
// W1t/W2t are transposed bf16 weights: Wt[n*128 + k] = W[k,n].
// One block = 16 rows x full 128 cols; 8 waves, one 16x16 tile per wave.

__device__ __forceinline__ v16bf load_a_frag(const bf16_t* __restrict__ base,
                                             int row0, int k0, int lane) {
    // 16-bit A layout: lanes 0-15 row M=lane, elems = K{k0..k0+7, k0+16..k0+23}
    //                  lanes 16-31 same rows, +8 shifted K set
    const bf16_t* p = base + (size_t)(row0 + (lane & 15)) * HDIM + k0 + ((lane >> 4) << 3);
    v8bf lo = *(const v8bf*)p;
    v8bf hi = *(const v8bf*)(p + 16);
    v16bf f;
#pragma unroll
    for (int i = 0; i < 8; ++i) { f[i] = lo[i]; f[i + 8] = hi[i]; }
    return f;
}

__device__ __forceinline__ v16bf load_b_frag(const bf16_t* __restrict__ wt,
                                             int col0, int k0, int lane) {
    // B (KxN): lane holds column n = lane&15; lanes 0-15 K={k0..k0+7, k0+16..k0+23},
    // lanes 16-31 same with +8 shifted K set (matches A fragment K order)
    const bf16_t* p = wt + (size_t)(col0 + (lane & 15)) * HDIM + k0 + ((lane >> 4) << 3);
    v8bf lo = *(const v8bf*)p;
    v8bf hi = *(const v8bf*)(p + 16);
    v16bf f;
#pragma unroll
    for (int i = 0; i < 8; ++i) { f[i] = lo[i]; f[i + 8] = hi[i]; }
    return f;
}

__global__ __launch_bounds__(256) void k_gemm_dual(const bf16_t* __restrict__ A1,
                                                   const bf16_t* __restrict__ W1t,
                                                   const bf16_t* __restrict__ A2,
                                                   const bf16_t* __restrict__ W2t,
                                                   const float* __restrict__ bias,
                                                   bf16_t* __restrict__ out) {
    const int lane = threadIdx.x & 31;
    const int wave = threadIdx.x >> 5;
    const int row0 = blockIdx.x * 16;
    const int col0 = wave * 16;

    v8f acc = (v8f)0.0f;
#pragma unroll
    for (int kt = 0; kt < HDIM; kt += 32) {
        v16bf a1 = load_a_frag(A1, row0, kt, lane);
        v16bf b1 = load_b_frag(W1t, col0, kt, lane);
        acc = __builtin_amdgcn_wmma_f32_16x16x32_bf16(false, a1, false, b1,
                                                      (short)0, acc, false, false);
        v16bf a2 = load_a_frag(A2, row0, kt, lane);
        v16bf b2 = load_b_frag(W2t, col0, kt, lane);
        acc = __builtin_amdgcn_wmma_f32_16x16x32_bf16(false, a2, false, b2,
                                                      (short)0, acc, false, false);
    }

    // C/D layout: VGPR r -> row = row0 + r + 8*(lane>=16), col = col0 + (lane&15)
    const int n = col0 + (lane & 15);
    const float bv = bias[n];
    const int rbase = row0 + ((lane >> 4) << 3);
#pragma unroll
    for (int r = 0; r < 8; ++r) {
        float v = acc[r] + bv;
        v = fmaxf(v, 0.0f);
        out[(size_t)(rbase + r) * HDIM + n] = (bf16_t)v;
    }
}

// ---------------------------------------------------------------- head
// per-graph: BN(eval) -> relu(g@lin1 + b1) -> g@lin2 + b2 -> softmax
__global__ __launch_bounds__(128) void k_head(const float* __restrict__ pool,
                                              const float* __restrict__ gamma,
                                              const float* __restrict__ beta,
                                              const float* __restrict__ rm,
                                              const float* __restrict__ rv,
                                              const float* __restrict__ W1,
                                              const float* __restrict__ b1,
                                              const float* __restrict__ W2,
                                              const float* __restrict__ b2,
                                              float* __restrict__ out) {
    __shared__ float g[2 * HDIM];
    __shared__ float hv[HDIM];
    __shared__ float lg[NCLS];
    const int gi = blockIdx.x;
    const int t = threadIdx.x;

    for (int j = t; j < 2 * HDIM; j += 128) {
        float v = pool[(size_t)gi * (2 * HDIM) + j];
        g[j] = (v - rm[j]) * rsqrtf(rv[j] + 1e-5f) * gamma[j] + beta[j];
    }
    __syncthreads();

    float acc = b1[t];
    for (int k = 0; k < 2 * HDIM; ++k) acc += g[k] * W1[k * HDIM + t];
    hv[t] = fmaxf(acc, 0.0f);
    __syncthreads();

    if (t < NCLS) {
        float a = b2[t];
        for (int k = 0; k < HDIM; ++k) a += hv[k] * W2[k * NCLS + t];
        lg[t] = a;
    }
    __syncthreads();

    if (t == 0) {
        float m = lg[0];
        for (int c = 1; c < NCLS; ++c) m = fmaxf(m, lg[c]);
        float s = 0.0f, e[NCLS];
        for (int c = 0; c < NCLS; ++c) { e[c] = expf(lg[c] - m); s += e[c]; }
        float inv = 1.0f / s;
        for (int c = 0; c < NCLS; ++c) out[(size_t)gi * NCLS + c] = e[c] * inv;
    }
}

// ---------------------------------------------------------------- driver

extern "C" void kernel_launch(void* const* d_in, const int* in_sizes, int n_in,
                              void* d_out, int out_size, void* d_ws, size_t ws_size,
                              hipStream_t stream) {
    (void)in_sizes; (void)n_in; (void)out_size; (void)ws_size;

    const float* x     = (const float*)d_in[0];
    const int*   ei    = (const int*)d_in[1];
    const int*   src   = ei;
    const int*   dst   = ei + N_EDGES;
    const int*   batch = (const int*)d_in[2];

    const float* bW[2][8];
    for (int b = 0; b < 2; ++b)
        for (int j = 0; j < 8; ++j)
            bW[b][j] = (const float*)d_in[4 + b * 8 + j];
    // order per block: Wl1, Wr1, b1, Wl2, Wr2, b2, Wlin, blin

    const float* bn_gamma = (const float*)d_in[20];
    const float* bn_beta  = (const float*)d_in[21];
    const float* bn_rm    = (const float*)d_in[22];
    const float* bn_rv    = (const float*)d_in[23];
    const float* lin1_W   = (const float*)d_in[24];
    const float* lin1_b   = (const float*)d_in[25];
    const float* lin2_W   = (const float*)d_in[26];
    const float* lin2_b   = (const float*)d_in[27];

    // ---- workspace carve-out (256B aligned)
    size_t off = 0;
    char* wsb = (char*)d_ws;
    auto take = [&](size_t bytes) -> void* {
        void* p = wsb + off;
        off += (bytes + 255) & ~(size_t)255;
        return p;
    };
    float*  deg   = (float*)take((size_t)N_NODES * 4);
    float*  agg   = (float*)take((size_t)N_NODES * HDIM * 4);
    bf16_t* actx  = (bf16_t*)take((size_t)N_NODES * HDIM * 2);
    bf16_t* meanb = (bf16_t*)take((size_t)N_NODES * HDIM * 2);
    bf16_t* h1b   = (bf16_t*)take((size_t)N_NODES * HDIM * 2);
    bf16_t* h2b   = (bf16_t*)take((size_t)N_NODES * HDIM * 2);
    bf16_t* hbb   = (bf16_t*)take((size_t)N_NODES * HDIM * 2);
    float*  pool  = (float*)take((size_t)N_GRAPH * 2 * HDIM * 4);
    // transposed bf16 weights: per block {Wl1,Wr1,Wl2,Wr2,WlinTop,WlinBot}
    bf16_t* wt[2][6];
    for (int b = 0; b < 2; ++b)
        for (int j = 0; j < 6; ++j)
            wt[b][j] = (bf16_t*)take((size_t)HDIM * HDIM * 2);

    const int NB_NH  = (N_NODES * HDIM) / 256;  // 25000
    const int NB_TR  = (HDIM * HDIM) / 256;     // 64
    const int NB_DEG = N_EDGES / 256;           // 6250
    const int NB_SC  = (N_EDGES * 32) / 256;    // 200000
    const int NB_GE  = N_NODES / 16;            // 3125

    hipMemsetAsync(deg, 0, (size_t)N_NODES * 4, stream);
    hipMemsetAsync(pool, 0, (size_t)N_GRAPH * 2 * HDIM * 4, stream);

    k_degree<<<NB_DEG, 256, 0, stream>>>(dst, deg);
    k_f32_to_bf16<<<NB_NH, 256, 0, stream>>>(x, actx);

    for (int b = 0; b < 2; ++b) {
        k_transpose128<<<NB_TR, 256, 0, stream>>>(bW[b][0], wt[b][0]);            // Wl1
        k_transpose128<<<NB_TR, 256, 0, stream>>>(bW[b][1], wt[b][1]);            // Wr1
        k_transpose128<<<NB_TR, 256, 0, stream>>>(bW[b][3], wt[b][2]);            // Wl2
        k_transpose128<<<NB_TR, 256, 0, stream>>>(bW[b][4], wt[b][3]);            // Wr2
        k_transpose128<<<NB_TR, 256, 0, stream>>>(bW[b][6], wt[b][4]);            // Wlin rows 0..127
        k_transpose128<<<NB_TR, 256, 0, stream>>>(bW[b][6] + HDIM * HDIM, wt[b][5]); // Wlin rows 128..255
    }

    const bf16_t* act = actx;
    for (int b = 0; b < 2; ++b) {
        // conv1: h1 = relu(mean@Wl1 + act@Wr1 + b1)
        hipMemsetAsync(agg, 0, (size_t)N_NODES * HDIM * 4, stream);
        k_scatter<<<NB_SC, 256, 0, stream>>>(act, src, dst, agg);
        k_mean_cvt<<<NB_NH, 256, 0, stream>>>(agg, deg, meanb);
        k_gemm_dual<<<NB_GE, 256, 0, stream>>>(meanb, wt[b][0], act, wt[b][1], bW[b][2], h1b);

        // conv2: h2 = relu(mean@Wl2 + h1@Wr2 + b2)
        hipMemsetAsync(agg, 0, (size_t)N_NODES * HDIM * 4, stream);
        k_scatter<<<NB_SC, 256, 0, stream>>>(h1b, src, dst, agg);
        k_mean_cvt<<<NB_NH, 256, 0, stream>>>(agg, deg, meanb);
        k_gemm_dual<<<NB_GE, 256, 0, stream>>>(meanb, wt[b][2], h1b, wt[b][3], bW[b][5], h2b);

        // JK cat + linear: hb = relu(h1@Wlin[:128] + h2@Wlin[128:] + blin)
        k_gemm_dual<<<NB_GE, 256, 0, stream>>>(h1b, wt[b][4], h2b, wt[b][5], bW[b][7], hbb);

        // global_add_pool into pool[:, b*128 : b*128+128]
        k_pool<<<NB_NH, 256, 0, stream>>>(hbb, batch, pool, b * HDIM);

        act = hbb;  // block 1 consumes block 0 output
    }

    k_head<<<N_GRAPH, 128, 0, stream>>>(pool, bn_gamma, bn_beta, bn_rm, bn_rv,
                                        lin1_W, lin1_b, lin2_W, lin2_b, (float*)d_out);
}
